// ResidualStack_45183055954109
// MI455X (gfx1250) — compile-verified, matrix operand order checked by reading, not run on hardware
//
#include <hip/hip_runtime.h>
#include <hip/hip_bf16.h>

// ---------------------------------------------------------------------------
// WaveNet-style residual stack for MI455X (gfx1250, wave32).
// Per block: H = relu(W0 @ X[:, :-d] + W1 @ X[:, d:] + b)
//            out = Wc @ H + bc + X[:, d:]
//            skip = (Ws @ H + bs)[:, -4096:]
// All matmuls are M x 64 GEMMs over timesteps -> V_WMMA_F32_16X16X32_BF16,
// f32 accumulation, f32 residual path kept in workspace ping-pong buffers.
// ---------------------------------------------------------------------------

typedef __attribute__((ext_vector_type(16))) __bf16 v16bf;
typedef __attribute__((ext_vector_type(8)))  __bf16 v8bf;
typedef __attribute__((ext_vector_type(8)))  float  v8f;

#define C_CH   64
#define SC_CH  256
#define BATCH  4
#define NBLK   40
#define SKIP   4096
#define TN     128            // timesteps per workgroup tile
#define KPAD   72             // padded row stride (halves) to dodge bank conflicts

// Build a 16-element bf16 WMMA fragment from one LDS row.
// ISA 7.12.2 (16-bit A 16x32): lane<16 -> K pairs {base+0..7, base+16..23},
// lane>=16 -> base+8.  Both chunks are 16B-aligned contiguous -> ds_load_b128.
__device__ __forceinline__ v16bf frag_row(const __bf16* row, int kbase) {
    union { v16bf v; v8bf h[2]; } u;
    u.h[0] = *(const v8bf*)(row + kbase);
    u.h[1] = *(const v8bf*)(row + kbase + 16);
    return u.v;
}

__device__ __forceinline__ v8f wmma_bf16(v16bf a, v16bf b, v8f c) {
    return __builtin_amdgcn_wmma_f32_16x16x32_bf16(
        /*neg_a=*/false, a, /*neg_b=*/false, b,
        /*c_mod=*/(short)0, c, /*reuse_a=*/false, /*reuse_b=*/false);
}

__global__ __launch_bounds__(256)
void ResidualStack_45183055954109_kernel(
    const float* __restrict__ x_in,   // [B, C, L_in]  f32
    float*       __restrict__ x_out,  // [B, C, L_out] f32
    const float* __restrict__ dc_w,   // [NB, C, C, 2]
    const float* __restrict__ dc_b,   // [NB, C]
    const float* __restrict__ conv_w, // [NB, C, C]
    const float* __restrict__ conv_b, // [NB, C]
    const float* __restrict__ skip_w, // [NB, SC, C]
    const float* __restrict__ skip_b, // [NB, SC]
    float*       __restrict__ skips,  // [NB, B, SC, SKIP]
    int blk, int L_in, int d)
{
    const int L_out   = L_in - d;
    const int b       = blockIdx.y;
    const int tile_t0 = blockIdx.x * TN;
    const int tid     = threadIdx.x;

    // ---- LDS staging (bf16 operands, f32 biases) --------------------------
    __shared__ __align__(16) __bf16 sW0[C_CH  * KPAD];   // dilated tap 0  [o][i]
    __shared__ __align__(16) __bf16 sW1[C_CH  * KPAD];   // dilated tap 1  [o][i]
    __shared__ __align__(16) __bf16 sWc[C_CH  * KPAD];   // residual conv  [o][i]
    __shared__ __align__(16) __bf16 sWs[SC_CH * KPAD];   // skip proj      [o][i]
    __shared__ __align__(16) __bf16 sXa[TN * KPAD];      // x[:, t]   [t][c]
    __shared__ __align__(16) __bf16 sXb[TN * KPAD];      // x[:, t+d] [t][c]
    __shared__ __align__(16) __bf16 sG [TN * KPAD];      // relu(H)   [t][c]
    __shared__ float sBdc[C_CH], sBc[C_CH], sBs[SC_CH];

    // ---- cooperative weight load (f32 global -> bf16 LDS) -----------------
    {
        const float* w0base = dc_w + (size_t)blk * C_CH * C_CH * 2;
        const float* wcbase = conv_w + (size_t)blk * C_CH * C_CH;
        for (int idx = tid; idx < C_CH * C_CH; idx += 256) {
            int o = idx >> 6, ci = idx & 63;
            const float* w = w0base + (size_t)(o * C_CH + ci) * 2;
            sW0[o * KPAD + ci] = (__bf16)w[0];
            sW1[o * KPAD + ci] = (__bf16)w[1];
            sWc[o * KPAD + ci] = (__bf16)wcbase[idx];
        }
        const float* wsbase = skip_w + (size_t)blk * SC_CH * C_CH;
        for (int idx = tid; idx < SC_CH * C_CH; idx += 256) {
            int o = idx >> 6, ci = idx & 63;
            sWs[o * KPAD + ci] = (__bf16)wsbase[idx];
        }
        if (tid < C_CH) {
            sBdc[tid] = dc_b[(size_t)blk * C_CH + tid];
            sBc[tid]  = conv_b[(size_t)blk * C_CH + tid];
        }
        if (tid < SC_CH) sBs[tid] = skip_b[(size_t)blk * SC_CH + tid];
    }

    // ---- activation tile load: coalesced along t, transpose to [t][c] -----
    const float* xb_base = x_in + (size_t)b * C_CH * L_in;
    for (int idx = tid; idx < C_CH * TN; idx += 256) {
        int j = idx & (TN - 1);      // timestep within tile (fast -> coalesced)
        int c = idx >> 7;            // channel
        int t = tile_t0 + j;
        if (t > L_out - 1) t = L_out - 1;            // edge clamp (results masked)
        sXa[j * KPAD + c] = (__bf16)xb_base[(size_t)c * L_in + t];
        sXb[j * KPAD + c] = (__bf16)xb_base[(size_t)c * L_in + t + d];
    }
    __syncthreads();

    // ---- per-wave WMMA compute: wave wv owns 16 timesteps -----------------
    const int lane  = tid & 31;
    const int wv    = tid >> 5;
    const int n     = lane & 15;                 // N index within 16-tile
    const int kb    = (lane >= 16) ? 8 : 0;      // half-lane K offset
    const int trow  = wv * 16 + n;               // timestep row in LDS tile
    const int tglob = tile_t0 + trow;            // global output timestep

    const __bf16* xaRow = &sXa[trow * KPAD];
    const __bf16* xbRow = &sXb[trow * KPAD];
    const __bf16* gRow  = &sG [trow * KPAD];

    v16bf bXa[2], bXb[2];
#pragma unroll
    for (int kk = 0; kk < 2; kk++) {
        bXa[kk] = frag_row(xaRow, kk * 32 + kb);
        bXb[kk] = frag_row(xbRow, kk * 32 + kb);
    }

    // H = W0@Xa + W1@Xb, bias+relu, bf16 G back to LDS (same-wave reuse).
#pragma unroll
    for (int mt = 0; mt < 4; mt++) {
        v8f acc = {0.f, 0.f, 0.f, 0.f, 0.f, 0.f, 0.f, 0.f};
        const __bf16* r0 = &sW0[(mt * 16 + n) * KPAD];
        const __bf16* r1 = &sW1[(mt * 16 + n) * KPAD];
#pragma unroll
        for (int kk = 0; kk < 2; kk++) {
            acc = wmma_bf16(frag_row(r0, kk * 32 + kb), bXa[kk], acc);
            acc = wmma_bf16(frag_row(r1, kk * 32 + kb), bXb[kk], acc);
        }
        v8bf g;
#pragma unroll
        for (int r = 0; r < 8; r++) {
            int m = mt * 16 + kb + r;            // C/D layout: lane>=16 -> M+8
            float v = acc[r] + sBdc[m];
            g[r] = (__bf16)(v > 0.f ? v : 0.f);
        }
        *(v8bf*)&sG[trow * KPAD + mt * 16 + kb] = g;  // contiguous 16B store
    }
    // This wave wrote all 64 channels of its own 16 timesteps; LDS ops from
    // the same wave are in-order, so no workgroup barrier is needed here.

    v16bf bG[2];
#pragma unroll
    for (int kk = 0; kk < 2; kk++) bG[kk] = frag_row(gRow, kk * 32 + kb);

    // R = Wc@G + bc + residual (f32 path), store next activation buffer.
#pragma unroll
    for (int mt = 0; mt < 4; mt++) {
        v8f acc = {0.f, 0.f, 0.f, 0.f, 0.f, 0.f, 0.f, 0.f};
        const __bf16* rw = &sWc[(mt * 16 + n) * KPAD];
#pragma unroll
        for (int kk = 0; kk < 2; kk++)
            acc = wmma_bf16(frag_row(rw, kk * 32 + kb), bG[kk], acc);
        if (tglob < L_out) {
#pragma unroll
            for (int r = 0; r < 8; r++) {
                int m = mt * 16 + kb + r;
                float v = acc[r] + sBc[m]
                        + xb_base[(size_t)m * L_in + tglob + d];  // f32 residual
                x_out[((size_t)b * C_CH + m) * L_out + tglob] = v;
            }
        }
    }

    // S = Ws@G + bs for the last-4096 window only (wave-uniform predicate).
    const int skip_start = L_out - SKIP;
    if (tile_t0 + wv * 16 + 16 > skip_start) {
#pragma unroll
        for (int mt = 0; mt < 16; mt++) {
            v8f acc = {0.f, 0.f, 0.f, 0.f, 0.f, 0.f, 0.f, 0.f};
            const __bf16* rw = &sWs[(mt * 16 + n) * KPAD];
#pragma unroll
            for (int kk = 0; kk < 2; kk++)
                acc = wmma_bf16(frag_row(rw, kk * 32 + kb), bG[kk], acc);
            int col = tglob - skip_start;
            if (col >= 0 && tglob < L_out) {
#pragma unroll
                for (int r = 0; r < 8; r++) {
                    int m = mt * 16 + kb + r;
                    skips[(((size_t)blk * BATCH + b) * SC_CH + m) * SKIP + col]
                        = acc[r] + sBs[m];
                }
            }
        }
    }
}

extern "C" void kernel_launch(void* const* d_in, const int* in_sizes, int n_in,
                              void* d_out, int out_size, void* d_ws, size_t ws_size,
                              hipStream_t stream) {
    const float* x      = (const float*)d_in[0];
    const float* dc_w   = (const float*)d_in[1];
    const float* dc_b   = (const float*)d_in[2];
    const float* conv_w = (const float*)d_in[3];
    const float* conv_b = (const float*)d_in[4];
    const float* skip_w = (const float*)d_in[5];
    const float* skip_b = (const float*)d_in[6];
    float* skips = (float*)d_out;

    // Ping-pong f32 activation buffers in workspace: 2 x (4*64*8192) floats = 16 MB.
    float* bufA = (float*)d_ws;
    float* bufB = bufA + (size_t)BATCH * C_CH * 8192;

    int dil[NBLK];
    {
        int k = 0;
        for (int s = 0; s < 4; s++)
            for (int l = 0; l < 10; l++) dil[k++] = 1 << l;
    }

    const float* cur = x;
    float* nxt = bufA;
    int L = 8192;
    for (int i = 0; i < NBLK; i++) {
        int dd = dil[i];
        int Lo = L - dd;
        dim3 grid((Lo + TN - 1) / TN, BATCH);
        ResidualStack_45183055954109_kernel<<<grid, dim3(256), 0, stream>>>(
            cur, nxt, dc_w, dc_b, conv_w, conv_b, skip_w, skip_b,
            skips, i, L, dd);
        cur = nxt;
        nxt = (nxt == bufA) ? bufB : bufA;
        L = Lo;
    }
}
